// AutoregressiveDecoder_16054587752583
// MI455X (gfx1250) — compile-verified, matrix operand order checked by reading, not run on hardware
//
#include <hip/hip_runtime.h>
#include <hip/hip_bf16.h>
#include <stdint.h>

// ---- problem constants (match reference) -----------------------------------
#define HID   1024
#define BATCH 64
#define NLAY  3
#define NTOK  100          // start symbol index = NTOK
#define UMAX  256
#define STEPS (UMAX + 1)   // 257
#define OUTD  1024

// LDS tile row stride: 1024 elems + 8 pad elems (16B) -> 2064B = 516 DWORDs.
// Row r starts at bank (4*r)%64, so the 16 lanes of a ds_load_b128 (one row
// each, same column) hit disjoint 4-bank groups: conflict-free.
#define LDSR  1032

typedef __bf16 bf16;
typedef __bf16 v16bf __attribute__((ext_vector_type(16)));
typedef __bf16 v8bf  __attribute__((ext_vector_type(8)));
typedef float  v8f   __attribute__((ext_vector_type(8)));
typedef unsigned int v4u __attribute__((ext_vector_type(4)));
typedef int v8i __attribute__((ext_vector_type(8)));
typedef int v4i __attribute__((ext_vector_type(4)));

static __device__ __forceinline__ v16bf cat8(v8bf lo, v8bf hi) {
    return __builtin_shufflevector(lo, hi, 0, 1, 2, 3, 4, 5, 6, 7,
                                           8, 9, 10, 11, 12, 13, 14, 15);
}

// 2x 16B loads (global_load_b128 or ds_load_b128 depending on pointer origin)
static __device__ __forceinline__ v16bf load_frag(const bf16* __restrict__ row, int kb) {
    v8bf lo = *reinterpret_cast<const v8bf*>(row + kb);
    v8bf hi = *reinterpret_cast<const v8bf*>(row + kb + 16);
    return cat8(lo, hi);
}

static __device__ __forceinline__ float sigmoidf_(float x) {
    return 1.0f / (1.0f + __expf(-x));
}

// ---- TDM staging: 16 rows x 1024 bf16, global row stride 1024 elems,
// ---- LDS row stride LDSR elems (via TDM iterate mode). Issue from ONE wave.
static __device__ __forceinline__ void tdm_stage_tile(bf16* lds_ptr, unsigned lds_off,
                                                      const bf16* g) {
#if __has_builtin(__builtin_amdgcn_tensor_load_to_lds)
    (void)lds_ptr;
    unsigned long long ga = (unsigned long long)(uintptr_t)g;
    // D# group0: count=1 (user), lds_addr, global_addr[56:0], type=2 ("image")
    v4u g0 = { 1u,
               lds_off,
               (unsigned)(ga & 0xFFFFFFFFu),
               (unsigned)((ga >> 32) & 0x01FFFFFFu) | (2u << 30) };
    // D# group1: data_size=1 (2B), iterate_enable=1; tensor 1024x1, tile 1024x1,
    // dim0 stride 1024 elems. (bit fields per CDNA5 ISA)
    v8i g1;
    g1[0] = (int)((1u << 16) | (1u << 19));   // data_size | iterate_enable
    g1[1] = (int)((unsigned)HID << 16);       // tensor_dim0[15:0] in [63:48]
    g1[2] = (int)(1u << 16);                  // tensor_dim0 hi=0 | tensor_dim1=1
    g1[3] = (int)((unsigned)HID << 16);       // tensor_dim1 hi=0 | tile_dim0=1024
    g1[4] = 1;                                // tile_dim1=1 (one row per iter)
    g1[5] = HID;                              // tensor_dim0_stride lo32
    g1[6] = 0;                                // stride hi | dim1_stride lo
    g1[7] = 0;                                // dim1_stride hi
    // D# group2 (iterate mode): lds_addr_increment=LDSR, global_addr_increment=HID,
    // iterate_count=15 -> 16 iterations (16 rows)
    v4i g2 = { 0, (int)LDSR, (int)HID, (int)(15u << 16) };
    v4i g3 = { 0, 0, 0, 0 };
    v8i g4 = { 0, 0, 0, 0, 0, 0, 0, 0 };      // unused trailing group (6-arg form)
    __builtin_amdgcn_tensor_load_to_lds(g0, g1, g2, g3, g4, 0);
#else
    // Fallback: cooperative copy by this wave (32 lanes), 16B per lane-step.
    (void)lds_off;
    const int lane = threadIdx.x & 31;
    for (int r = 0; r < 16; ++r)
        for (int c = lane * 8; c < HID; c += 32 * 8)
            *reinterpret_cast<v8bf*>(lds_ptr + r * LDSR + c) =
                *reinterpret_cast<const v8bf*>(g + r * HID + c);
#endif
}

// ---- prep kernels ----------------------------------------------------------
__global__ void cvt_f32_bf16(const float* __restrict__ src, bf16* __restrict__ dst, int n) {
    int i = blockIdx.x * blockDim.x + threadIdx.x;
    int stride = gridDim.x * blockDim.x;
    for (; i < n; i += stride) dst[i] = (bf16)src[i];
}

__global__ void init_state_kernel(const float* __restrict__ init_state,  // [L,H]
                                  float* __restrict__ h_f32,             // [L,B,H]
                                  bf16*  __restrict__ h_b16) {
    int i = blockIdx.x * blockDim.x + threadIdx.x;
    if (i >= NLAY * BATCH * HID) return;
    int l = i / (BATCH * HID);
    int h = i & (HID - 1);
    float v = init_state[l * HID + h];
    h_f32[i] = v;
    h_b16[i] = (bf16)v;
}

__global__ void embed_gather_kernel(const int* __restrict__ y,        // [B,UMAX]
                                    const bf16* __restrict__ emb,     // [NTOK+1,H] bf16
                                    bf16* __restrict__ x0,            // [B,H]
                                    int u) {
    int i = blockIdx.x * blockDim.x + threadIdx.x;
    if (i >= BATCH * HID) return;
    int b = i >> 10;
    int h = i & (HID - 1);
    int tok = (u == 0) ? NTOK : y[b * UMAX + (u - 1)];
    x0[i] = emb[tok * HID + h];
}

// ---- fused GRU layer step --------------------------------------------------
// grid = (16, 4): blockIdx.x = group of 4 N-tiles, blockIdx.y = M tile (16 rows)
// block = 128 (4 waves). Waves 0/1 TDM-stage the x-tile / h-tile into LDS
// (A-fragments are shared by all 4 waves: same M tile). Each wave then runs
// 6 bf16 WMMA accumulations (r/z/n x input/hidden) over K=1024 with A from
// LDS and B (weights, L2-resident) from global, and applies the GRU gates.
__launch_bounds__(128)
__global__ void gru_step_kernel(const bf16*  __restrict__ x,        // [B,H] layer input (bf16)
                                const float* __restrict__ hprev_f,  // [B,H]
                                const bf16*  __restrict__ hprev_b,  // [B,H]
                                const bf16*  __restrict__ Wih,      // [3H,H]
                                const bf16*  __restrict__ Whh,      // [3H,H]
                                const float* __restrict__ bih,      // [3H]
                                const float* __restrict__ bhh,      // [3H]
                                float* __restrict__ hnew_f,         // [B,H]
                                bf16*  __restrict__ hnew_b) {
    __shared__ __attribute__((aligned(16))) bf16 sx[16 * LDSR];
    __shared__ __attribute__((aligned(16))) bf16 sh[16 * LDSR];

    const int lane = threadIdx.x & 31;
    const int wave = threadIdx.x >> 5;
    const int lhi  = (lane >> 4) & 1;   // lane half: selects K sub-blocks
    const int lmod = lane & 15;
    const int m0 = blockIdx.y * 16;
    const int n0 = (blockIdx.x * 4 + wave) * 16;

    if (wave == 0) {
        tdm_stage_tile(sx, (unsigned)(uintptr_t)(void*)sx, x + (size_t)m0 * HID);
        __builtin_amdgcn_s_wait_tensorcnt(0);
    } else if (wave == 1) {
        tdm_stage_tile(sh, (unsigned)(uintptr_t)(void*)sh, hprev_b + (size_t)m0 * HID);
        __builtin_amdgcn_s_wait_tensorcnt(0);
    }
    __syncthreads();

    v8f acc_ir = {}, acc_iz = {}, acc_in = {};
    v8f acc_hr = {}, acc_hz = {}, acc_hn = {};

    const bf16* xrow = sx + lmod * LDSR;
    const bf16* hrow = sh + lmod * LDSR;
    const bf16* wir  = Wih + (size_t)(0 * HID + n0 + lmod) * HID;
    const bf16* wiz  = Wih + (size_t)(1 * HID + n0 + lmod) * HID;
    const bf16* win  = Wih + (size_t)(2 * HID + n0 + lmod) * HID;
    const bf16* whr  = Whh + (size_t)(0 * HID + n0 + lmod) * HID;
    const bf16* whz  = Whh + (size_t)(1 * HID + n0 + lmod) * HID;
    const bf16* whn  = Whh + (size_t)(2 * HID + n0 + lmod) * HID;

#pragma unroll 2
    for (int k0 = 0; k0 < HID; k0 += 32) {
        const int kb = k0 + lhi * 8;
        // speculative WGP-scope prefetch of next K-tile of L2-resident weights
        __builtin_prefetch(wir + kb + 64, 0, 3);
        __builtin_prefetch(whr + kb + 64, 0, 3);

        v16bf ax = load_frag(xrow, kb);   // ds_load_b128 x2
        v16bf ah = load_frag(hrow, kb);

        v16bf b0 = load_frag(wir, kb);
        acc_ir = __builtin_amdgcn_wmma_f32_16x16x32_bf16(false, ax, false, b0, (short)0, acc_ir, false, false);
        v16bf b1 = load_frag(wiz, kb);
        acc_iz = __builtin_amdgcn_wmma_f32_16x16x32_bf16(false, ax, false, b1, (short)0, acc_iz, false, false);
        v16bf b2 = load_frag(win, kb);
        acc_in = __builtin_amdgcn_wmma_f32_16x16x32_bf16(false, ax, false, b2, (short)0, acc_in, false, false);
        v16bf b3 = load_frag(whr, kb);
        acc_hr = __builtin_amdgcn_wmma_f32_16x16x32_bf16(false, ah, false, b3, (short)0, acc_hr, false, false);
        v16bf b4 = load_frag(whz, kb);
        acc_hz = __builtin_amdgcn_wmma_f32_16x16x32_bf16(false, ah, false, b4, (short)0, acc_hz, false, false);
        v16bf b5 = load_frag(whn, kb);
        acc_hn = __builtin_amdgcn_wmma_f32_16x16x32_bf16(false, ah, false, b5, (short)0, acc_hn, false, false);
    }

    // Fused GRU epilogue. C/D layout: elem e -> M = m0+e (+8 for lanes>=16), N = n0+lmod
    const int ncol = n0 + lmod;
    const float b_ir = bih[ncol], b_iz = bih[HID + ncol], b_in = bih[2 * HID + ncol];
    const float b_hr = bhh[ncol], b_hz = bhh[HID + ncol], b_hn = bhh[2 * HID + ncol];
#pragma unroll
    for (int e = 0; e < 8; ++e) {
        const int mrow = m0 + e + lhi * 8;
        const float r = sigmoidf_((acc_ir[e] + b_ir) + (acc_hr[e] + b_hr));
        const float z = sigmoidf_((acc_iz[e] + b_iz) + (acc_hz[e] + b_hz));
        const float n = tanhf((acc_in[e] + b_in) + r * (acc_hn[e] + b_hn));
        const float hold = hprev_f[(size_t)mrow * HID + ncol];
        const float hv = (1.0f - z) * n + z * hold;
        hnew_f[(size_t)mrow * HID + ncol] = hv;
        hnew_b[(size_t)mrow * HID + ncol] = (bf16)hv;
    }
}

// ---- joiner: out[:,u,:] = h2 @ lin_W^T + lin_b -----------------------------
__launch_bounds__(128)
__global__ void joiner_kernel(const bf16*  __restrict__ h2,     // [B,H] bf16
                              const bf16*  __restrict__ Wlin,   // [O,H] bf16
                              const float* __restrict__ blin,   // [O]
                              float* __restrict__ out,          // [B,STEPS,O]
                              int u) {
    __shared__ __attribute__((aligned(16))) bf16 sa[16 * LDSR];

    const int lane = threadIdx.x & 31;
    const int wave = threadIdx.x >> 5;
    const int lhi  = (lane >> 4) & 1;
    const int lmod = lane & 15;
    const int m0 = blockIdx.y * 16;
    const int n0 = (blockIdx.x * 4 + wave) * 16;

    if (wave == 0) {
        tdm_stage_tile(sa, (unsigned)(uintptr_t)(void*)sa, h2 + (size_t)m0 * HID);
        __builtin_amdgcn_s_wait_tensorcnt(0);
    }
    __syncthreads();

    v8f acc = {};
    const bf16* arow = sa + lmod * LDSR;
    const bf16* wrow = Wlin + (size_t)(n0 + lmod) * HID;

#pragma unroll 2
    for (int k0 = 0; k0 < HID; k0 += 32) {
        const int kb = k0 + lhi * 8;
        __builtin_prefetch(wrow + kb + 64, 0, 3);
        v16bf a = load_frag(arow, kb);
        v16bf b = load_frag(wrow, kb);
        acc = __builtin_amdgcn_wmma_f32_16x16x32_bf16(false, a, false, b, (short)0, acc, false, false);
    }

    const int ncol = n0 + lmod;
    const float bias = blin[ncol];
#pragma unroll
    for (int e = 0; e < 8; ++e) {
        const int mrow = m0 + e + lhi * 8;
        out[(size_t)mrow * (STEPS * OUTD) + (size_t)u * OUTD + ncol] = acc[e] + bias;
    }
}

// ---- launch ----------------------------------------------------------------
extern "C" void kernel_launch(void* const* d_in, const int* in_sizes, int n_in,
                              void* d_out, int out_size, void* d_ws, size_t ws_size,
                              hipStream_t stream) {
    (void)in_sizes; (void)n_in; (void)out_size; (void)ws_size;

    const int*   y        = (const int*)d_in[0];
    // d_in[1] = U (unused by reference math)
    const float* embed    = (const float*)d_in[2];
    const float* W_ih     = (const float*)d_in[3];
    const float* W_hh     = (const float*)d_in[4];
    const float* b_ih     = (const float*)d_in[5];
    const float* b_hh     = (const float*)d_in[6];
    const float* initS    = (const float*)d_in[7];
    const float* lin_W    = (const float*)d_in[8];
    const float* lin_bias = (const float*)d_in[9];
    float* out = (float*)d_out;

    // workspace carve-out (all 256B aligned)
    char*  ws  = (char*)d_ws;
    size_t off = 0;
    auto carve = [&](size_t bytes) -> void* {
        void* p = ws + off;
        off += (bytes + 255) & ~(size_t)255;
        return p;
    };
    bf16* Wih_b = (bf16*)carve((size_t)NLAY * 3 * HID * HID * sizeof(bf16));
    bf16* Whh_b = (bf16*)carve((size_t)NLAY * 3 * HID * HID * sizeof(bf16));
    bf16* Wln_b = (bf16*)carve((size_t)OUTD * HID * sizeof(bf16));
    bf16* emb_b = (bf16*)carve((size_t)(NTOK + 1) * HID * sizeof(bf16));
    bf16* x0    = (bf16*)carve((size_t)BATCH * HID * sizeof(bf16));
    bf16*  hb[2];
    float* hf[2];
    for (int p = 0; p < 2; ++p) {
        hb[p] = (bf16*) carve((size_t)NLAY * BATCH * HID * sizeof(bf16));
        hf[p] = (float*)carve((size_t)NLAY * BATCH * HID * sizeof(float));
    }

    // one-time weight conversion to bf16 (L2-resident afterwards: ~40 MB << 192 MB)
    cvt_f32_bf16<<<2048, 256, 0, stream>>>(W_ih, Wih_b, NLAY * 3 * HID * HID);
    cvt_f32_bf16<<<2048, 256, 0, stream>>>(W_hh, Whh_b, NLAY * 3 * HID * HID);
    cvt_f32_bf16<<<1024, 256, 0, stream>>>(lin_W, Wln_b, OUTD * HID);
    cvt_f32_bf16<<<256,  256, 0, stream>>>(embed, emb_b, (NTOK + 1) * HID);
    init_state_kernel<<<(NLAY * BATCH * HID + 255) / 256, 256, 0, stream>>>(initS, hf[0], hb[0]);

    const dim3 gemm_grid(16, 4);   // 64 N-tiles / 4 waves-per-WG  x  4 M-tiles
    const dim3 gemm_block(128);    // 4 waves (wave32)

    for (int u = 0; u < STEPS; ++u) {
        const int p = u & 1, q = p ^ 1;  // ping-pong state buffers (no intra-step RAW races)
        embed_gather_kernel<<<(BATCH * HID + 255) / 256, 256, 0, stream>>>(y, emb_b, x0, u);
        for (int l = 0; l < NLAY; ++l) {
            const bf16* xin = (l == 0) ? x0 : (hb[q] + (size_t)(l - 1) * BATCH * HID);
            gru_step_kernel<<<gemm_grid, gemm_block, 0, stream>>>(
                xin,
                hf[p] + (size_t)l * BATCH * HID,
                hb[p] + (size_t)l * BATCH * HID,
                Wih_b + (size_t)l * 3 * HID * HID,
                Whh_b + (size_t)l * 3 * HID * HID,
                b_ih + (size_t)l * 3 * HID,
                b_hh + (size_t)l * 3 * HID,
                hf[q] + (size_t)l * BATCH * HID,
                hb[q] + (size_t)l * BATCH * HID);
        }
        joiner_kernel<<<gemm_grid, gemm_block, 0, stream>>>(
            hb[q] + (size_t)2 * BATCH * HID, Wln_b, lin_bias, out, u);
    }
}
